// AttentionNet_86406152060989
// MI455X (gfx1250) — matrix operand order repaired and checked by reading, not compiled, not run on hardware
//
#include <hip/hip_runtime.h>
#include <hip/hip_bf16.h>

typedef _Float16 h8v  __attribute__((ext_vector_type(8)));
typedef _Float16 h16v __attribute__((ext_vector_type(16)));
typedef float    f8v  __attribute__((ext_vector_type(8)));
typedef unsigned int u32x4 __attribute__((ext_vector_type(4)));
typedef int          i32x4 __attribute__((ext_vector_type(4)));
typedef int          i32x8 __attribute__((ext_vector_type(8)));

// ---------------------------------------------------------------------------
// WMMA "NT" GEMM:  out[M,N] (f32) = A[M,K] (f16, row-major) @ Wt[N,K]^T + bias
// grid = (M/16, ceil(N/128), batch), block = 256 (8 waves, 1 tile/wave).
// The 16xK A tile is staged into LDS ONCE per block via the Tensor Data Mover
// (iterate mode: one 16x32-half tile per K-chunk, padded to 80B rows so the
// 16-lane ds_load_b128 column reads are bank-conflict-free), then all 8 waves
// read A from LDS. B is register-double-buffered against the WMMA.
// Dynamic LDS = (K/32) * 1280 bytes.
// ---------------------------------------------------------------------------
__global__ void wmma_gemm_nt(const _Float16* __restrict__ A, int lda, long long sA,
                             const _Float16* __restrict__ Wt, int ldw, long long sW,
                             const float* __restrict__ bias,
                             float* __restrict__ out, int ldo, long long sO,
                             int N, int K)
{
    extern __shared__ char dynsmem[];
    _Float16* lsA = (_Float16*)dynsmem;

    const int z = blockIdx.z;
    A   += (long long)z * sA;
    Wt  += (long long)z * sW;
    out += (long long)z * sO;
    const int wave  = threadIdx.x >> 5;
    const int lane  = threadIdx.x & 31;
    const int ntile = blockIdx.y * 8 + wave;
    const bool active = (ntile * 16 < N);
    const int m0  = blockIdx.x * 16;
    const int n0  = ntile * 16;
    const int hi  = lane >> 4;     // 0/1: K-half of the wave
    const int l15 = lane & 15;
    const int nchunks = K >> 5;

    // ---- stage A[m0:m0+16, 0:K] into LDS ----
#if defined(__has_builtin) && __has_builtin(__builtin_amdgcn_tensor_load_to_lds)
    if (wave == 0) {
        const unsigned lds_off = (unsigned)(unsigned long long)(void*)lsA;
        const unsigned long long ga =
            (unsigned long long)(const void*)(A + (long long)m0 * lda);
        u32x4 g0;
        g0[0] = 1u;                                   // count=1 (valid user D#)
        g0[1] = lds_off;                              // lds_addr
        g0[2] = (unsigned)(ga & 0xffffffffu);         // global_addr[31:0]
        g0[3] = (unsigned)((ga >> 32) & 0x01ffffffu)  // global_addr[56:32]
                | (2u << 30);                         // type = 2 ("image")
        i32x8 g1;
        g1[0] = (1 << 16)      // data_size = 2 bytes
              | (1 << 19)      // iterate_enable
              | (1 << 20)      // pad_enable
              | (3 << 22)      // pad_interval = 16 DWORDs (one 64B row)
              | (3 << 25);     // pad_amount   = 4 DWORDs (16B per row)
        g1[1] = (int)((lda & 0xffff) << 16);          // tensor_dim0[15:0]
        g1[2] = (int)((lda >> 16) & 0xffff)           // tensor_dim0[31:16]
              | (16 << 16);                           // tensor_dim1 = 16 rows
        g1[3] = (32 << 16);                           // tile_dim0 = 32 halves
        g1[4] = 16;                                   // tile_dim1 = 16 rows
        g1[5] = lda;                                  // tensor_dim0_stride
        g1[6] = 0;
        g1[7] = 0;
        i32x4 g2;
        g2[0] = 1;                                    // tensor_dim2 (unused)
        g2[1] = 640;                                  // lds_addr_increment (halves/chunk)
        g2[2] = 32;                                   // global_addr_increment (halves)
        g2[3] = (nchunks - 1) << 16;                  // iterate_count
        i32x4 g3 = {0, 0, 0, 0};
        i32x8 g4 = {0, 0, 0, 0, 0, 0, 0, 0};          // extra group (clang-23 6-arg form)
        __builtin_amdgcn_tensor_load_to_lds(g0, g1, g2, g3, g4, 0);
        __builtin_amdgcn_s_wait_tensorcnt(0);
    }
#else
    for (int idx = threadIdx.x; idx < (nchunks << 6); idx += blockDim.x) {
        const int c = idx >> 6;          // K-chunk
        const int r = (idx >> 2) & 15;   // row
        const int g = idx & 3;           // 8-half group
        *(h8v*)(lsA + c * 640 + r * 40 + g * 8) =
            *(const h8v*)(A + (long long)(m0 + r) * lda + c * 32 + g * 8);
    }
#endif
    __syncthreads();
    if (!active) return;

    // ---- main loop: A from LDS, B register-double-buffered from global ----
    const _Float16* bp = Wt + (long long)(n0 + l15) * ldw + hi * 8;
    const _Float16* arow = lsA + l15 * 40 + hi * 8;   // + chunk*640 per K-chunk
    h16v bv = __builtin_shufflevector(*(const h8v*)(bp), *(const h8v*)(bp + 16), 0,1,2,3,4,5,6,7,8,9,10,11,12,13,14,15);
    f8v acc = {};
    for (int k0 = 0; k0 < K; k0 += 32) {
        const h16v bcur = bv;
        if (k0 + 32 < K) {
            bv = __builtin_shufflevector(*(const h8v*)(bp + k0 + 32), *(const h8v*)(bp + k0 + 48), 0,1,2,3,4,5,6,7,8,9,10,11,12,13,14,15);
            __builtin_prefetch((const void*)(bp + k0 + 64), 0, 1);
        }
        const _Float16* ac = arow + (k0 >> 5) * 640;
        h8v alo = *(const h8v*)(ac);
        h8v ahi = *(const h8v*)(ac + 16);
        h16v av = __builtin_shufflevector(alo, ahi, 0,1,2,3,4,5,6,7,8,9,10,11,12,13,14,15);
        acc = __builtin_amdgcn_wmma_f32_16x16x32_f16(false, av, false, bcur,
                                                     (short)0, acc, false, false);
    }
    const int   mb = m0 + hi * 8;
    const int   nc = n0 + l15;
    const float bb = bias ? bias[nc] : 0.0f;
#pragma unroll
    for (int r = 0; r < 8; ++r)
        out[(long long)(mb + r) * ldo + nc] = acc[r] + bb;
}

// ---------------------------------------------------------------------------
// Fused GRU timestep: gh = h_in @ Whh^T + bhh via WMMA (6 waves = 2 M-tiles x
// 3 gates for one 16-wide hidden slice), then pointwise GRU update.
// grid = H/16 blocks, block = 192 threads. gi (input projection) precomputed.
// ---------------------------------------------------------------------------
__global__ void gru_step(const _Float16* __restrict__ h_in,   // (32, H) f16
                         _Float16* __restrict__ h_out,        // (32, H) f16
                         const _Float16* __restrict__ Whh,    // (3H, H) f16
                         const float* __restrict__ bhh,       // (3H)
                         const float* __restrict__ gi_t,      // + b*giStride -> (3H)
                         long long giStride,
                         float* __restrict__ outC,            // + b*outStride + j
                         long long outStride,
                         int H)
{
    __shared__ float ghs[3][32][17];
    const int tid  = threadIdx.x;
    const int wave = tid >> 5, lane = tid & 31;
    const int mtile = wave & 1, gate = wave >> 1;
    const int j0 = blockIdx.x * 16;
    const int n0 = gate * H + j0;
    const int hi = lane >> 4, l15 = lane & 15;
    const _Float16* ap = h_in + (long long)(mtile * 16 + l15) * H + hi * 8;
    const _Float16* bp = Whh  + (long long)(n0 + l15) * H + hi * 8;
    f8v acc = {};
    for (int k0 = 0; k0 < H; k0 += 32) {
        h8v alo = *(const h8v*)(ap + k0);
        h8v ahi = *(const h8v*)(ap + k0 + 16);
        h16v av = __builtin_shufflevector(alo, ahi, 0,1,2,3,4,5,6,7,8,9,10,11,12,13,14,15);
        h16v bv = __builtin_shufflevector(*(const h8v*)(bp + k0), *(const h8v*)(bp + k0 + 16), 0,1,2,3,4,5,6,7,8,9,10,11,12,13,14,15);
        acc = __builtin_amdgcn_wmma_f32_16x16x32_f16(false, av, false, bv,
                                                     (short)0, acc, false, false);
    }
    const int   mb = mtile * 16 + hi * 8;
    const float bb = bhh[n0 + l15];
#pragma unroll
    for (int r = 0; r < 8; ++r)
        ghs[gate][mb + r][l15] = acc[r] + bb;
    __syncthreads();
    for (int idx = tid; idx < 512; idx += blockDim.x) {
        const int b = idx >> 4, jj = idx & 15;
        const int j = j0 + jj;
        const float* gi = gi_t + (long long)b * giStride;
        float r  = gi[j]         + ghs[0][b][jj];
        float zz = gi[H + j]     + ghs[1][b][jj];
        float nn = gi[2 * H + j];
        const float ghn = ghs[2][b][jj];
        r  = 1.0f / (1.0f + __expf(-r));
        zz = 1.0f / (1.0f + __expf(-zz));
        nn = tanhf(nn + r * ghn);
        const float hp = (float)h_in[(long long)b * H + j];
        const float hv = (1.0f - zz) * nn + zz * hp;
        h_out[(long long)b * H + j] = (_Float16)hv;
        outC[(long long)b * outStride + j] = hv;
    }
}

// ------------------------------- elementwise -------------------------------
__global__ void cvt_f32_f16_k(const float* __restrict__ x, _Float16* __restrict__ y,
                              long long n)
{
    long long i = (long long)blockIdx.x * blockDim.x + threadIdx.x;
    if (i < n) y[i] = (_Float16)x[i];
}

__global__ void rowdot_k(const float* __restrict__ X, const float* __restrict__ w,
                         float* __restrict__ o, int rows, int D)
{
    int r = blockIdx.x * blockDim.x + threadIdx.x;
    if (r >= rows) return;
    const float* x = X + (long long)r * D;
    float s = 0.f;
    for (int d = 0; d < D; ++d) s += x[d] * w[d];
    o[r] = s;
}

__global__ void mulvec_cvt_k(const float* __restrict__ x, const float* __restrict__ w,
                             _Float16* __restrict__ y, long long n, int D)
{
    long long i = (long long)blockIdx.x * blockDim.x + threadIdx.x;
    if (i < n) y[i] = (_Float16)(x[i] * w[(int)(i % D)]);
}

__global__ void padq_cvt_k(const float* __restrict__ Q, _Float16* __restrict__ Qp,
                           int Jq, int D)
{   // Qp: (B, 32, D), rows >= Jq zero-filled
    long long i = (long long)blockIdx.x * blockDim.x + threadIdx.x;
    long long n = (long long)32 * 32 * D;
    if (i >= n) return;
    int d = (int)(i % D);
    int j = (int)((i / D) % 32);
    int b = (int)(i / ((long long)32 * D));
    Qp[i] = (j < Jq) ? (_Float16)Q[((long long)(b * Jq + j)) * D + d] : (_Float16)0.0f;
}

__global__ void transq_k(const _Float16* __restrict__ Qp, _Float16* __restrict__ QT, int D)
{   // QT: (B, D, 32) from Qp: (B, 32, D)
    long long i = (long long)blockIdx.x * blockDim.x + threadIdx.x;
    long long n = (long long)32 * D * 32;
    if (i >= n) return;
    int j = (int)(i % 32);
    int d = (int)((i / 32) % D);
    int b = (int)(i / ((long long)32 * D));
    QT[i] = Qp[((long long)b * 32 + j) * D + d];
}

__global__ void adds_k(float* __restrict__ S, const float* __restrict__ Cw1,
                       const float* __restrict__ Qw2, int T, int J)
{
    long long i = (long long)blockIdx.x * blockDim.x + threadIdx.x;
    long long n = (long long)32 * T * J;
    if (i >= n) return;
    int j = (int)(i % J);
    int t = (int)((i / J) % T);
    int b = (int)(i / ((long long)T * J));
    S[((long long)(b * T + t)) * 32 + j] += Cw1[b * T + t] + Qw2[b * J + j];
}

__global__ void attn_b_softmax_k(const float* __restrict__ S, float* __restrict__ bw,
                                 int T, int J)
{   // bw[b, :] = softmax over T of max_j S[b, :, j]
    const int b = blockIdx.x, tid = threadIdx.x;
    __shared__ float red[256];
    const float* Sb = S + (long long)b * T * 32;
    float gmax = -1e30f;
    for (int i = tid; i < T; i += blockDim.x) {
        float mi = -1e30f;
        for (int j = 0; j < J; ++j) mi = fmaxf(mi, Sb[(long long)i * 32 + j]);
        bw[(long long)b * T + i] = mi;
        gmax = fmaxf(gmax, mi);
    }
    red[tid] = gmax; __syncthreads();
    for (int s = 128; s > 0; s >>= 1) { if (tid < s) red[tid] = fmaxf(red[tid], red[tid + s]); __syncthreads(); }
    gmax = red[0]; __syncthreads();
    float sum = 0.f;
    for (int i = tid; i < T; i += blockDim.x) {
        float e = __expf(bw[(long long)b * T + i] - gmax);
        bw[(long long)b * T + i] = e;
        sum += e;
    }
    red[tid] = sum; __syncthreads();
    for (int s = 128; s > 0; s >>= 1) { if (tid < s) red[tid] += red[tid + s]; __syncthreads(); }
    const float inv = 1.0f / red[0]; __syncthreads();
    for (int i = tid; i < T; i += blockDim.x) bw[(long long)b * T + i] *= inv;
}

__global__ void q2c_k(const float* __restrict__ bw, const float* __restrict__ C,
                      float* __restrict__ q2c, int T, int D)
{
    const int b = blockIdx.x, d = threadIdx.x;   // blockDim == D (512)
    const float* cb = C + (long long)b * T * D + d;
    const float* wb = bw + (long long)b * T;
    float s = 0.f;
    for (int t = 0; t < T; ++t) s += wb[t] * cb[(long long)t * D];
    q2c[(long long)b * D + d] = s;
}

__global__ void build_g_k(const float* __restrict__ C, const float* __restrict__ c2q,
                          const float* __restrict__ q2c,
                          float* __restrict__ G, _Float16* __restrict__ G16,
                          int T, int D)
{   // G = [C, c2q, C*c2q, C*q2c], shape (B, T, 4D)
    long long i = (long long)blockIdx.x * blockDim.x + threadIdx.x;
    long long n = (long long)32 * T * 4 * D;
    if (i >= n) return;
    int k = (int)(i % (4 * D));
    long long row = i / (4 * D);           // b*T + t
    int b = (int)(row / T);
    int region = k / D, d = k % D;
    float cf = C[row * D + d];
    float v;
    if (region == 0)      v = cf;
    else if (region == 1) v = c2q[row * D + d];
    else if (region == 2) v = cf * c2q[row * D + d];
    else                  v = cf * q2c[(long long)b * D + d];
    G[i]   = v;
    G16[i] = (_Float16)v;
}

__global__ void gm_sum_k(const float* __restrict__ G, const float* __restrict__ Mx,
                         _Float16* __restrict__ o, int T, int DG, int DM)
{   // o[b, f] = sum_t concat(G, Mx)[b, t, f], f < DG+DM
    const int F = DG + DM;
    long long i = (long long)blockIdx.x * blockDim.x + threadIdx.x;
    if (i >= (long long)32 * F) return;
    int b = (int)(i / F), f = (int)(i % F);
    float s = 0.f;
    if (f < DG) {
        const float* g = G + (long long)b * T * DG + f;
        for (int t = 0; t < T; ++t) s += g[(long long)t * DG];
    } else {
        const float* m = Mx + (long long)b * T * DM + (f - DG);
        for (int t = 0; t < T; ++t) s += m[(long long)t * DM];
    }
    o[i] = (_Float16)s;
}

__global__ void softmax_rows_k(const float* __restrict__ x, float* __restrict__ o, int N)
{
    const int row = blockIdx.x, tid = threadIdx.x;
    __shared__ float red[256];
    const float* xr = x + (long long)row * N;
    float m = -1e30f;
    for (int i = tid; i < N; i += blockDim.x) m = fmaxf(m, xr[i]);
    red[tid] = m; __syncthreads();
    for (int s = 128; s > 0; s >>= 1) { if (tid < s) red[tid] = fmaxf(red[tid], red[tid + s]); __syncthreads(); }
    m = red[0]; __syncthreads();
    float sum = 0.f;
    for (int i = tid; i < N; i += blockDim.x) sum += __expf(xr[i] - m);
    red[tid] = sum; __syncthreads();
    for (int s = 128; s > 0; s >>= 1) { if (tid < s) red[tid] += red[tid + s]; __syncthreads(); }
    const float inv = 1.0f / red[0]; __syncthreads();
    for (int i = tid; i < N; i += blockDim.x) o[(long long)row * N + i] = __expf(xr[i] - m) * inv;
}

// ---------------------------------------------------------------------------
extern "C" void kernel_launch(void* const* d_in, const int* in_sizes, int n_in,
                              void* d_out, int out_size, void* d_ws, size_t ws_size,
                              hipStream_t stream)
{
    (void)in_sizes; (void)n_in; (void)out_size; (void)ws_size;
    enum { E2 = 256, E4 = 512, B = 32, T = 400, J = 30, D = 512,
           ANS = 400, G3 = 768, G3P = 1536, GF = 2048 };

    const float* embd_ctx = (const float*)d_in[0];
    const float* embd_q   = (const float*)d_in[1];
    const float* Wvec     = (const float*)d_in[2];
    const float* ctx_Wih  = (const float*)d_in[3];
    const float* ctx_Whh  = (const float*)d_in[4];
    const float* ctx_bih  = (const float*)d_in[5];
    const float* ctx_bhh  = (const float*)d_in[6];
    const float* mod_Wih  = (const float*)d_in[7];
    const float* mod_Whh  = (const float*)d_in[8];
    const float* mod_bih  = (const float*)d_in[9];
    const float* mod_bhh  = (const float*)d_in[10];
    const float* p2g_Wih  = (const float*)d_in[11];
    const float* p2g_Whh  = (const float*)d_in[12];
    const float* p2g_bih  = (const float*)d_in[13];
    const float* p2g_bhh  = (const float*)d_in[14];
    const float* p1_w     = (const float*)d_in[15];
    const float* p1_b     = (const float*)d_in[16];
    const float* p2_w     = (const float*)d_in[17];
    const float* p2_b     = (const float*)d_in[18];
    float* out = (float*)d_out;

    // ---- workspace bump allocator ----
    char* pw = (char*)d_ws;
    auto alloc = [&](size_t bytes) -> void* {
        void* r = (void*)pw;
        pw += (bytes + 255) & ~(size_t)255;
        return r;
    };
    _Float16* Xc16     = (_Float16*)alloc((size_t)B * T * E2 * 2);
    _Float16* Xq16     = (_Float16*)alloc((size_t)B * J * E2 * 2);
    _Float16* ctxWih16 = (_Float16*)alloc((size_t)2 * G3 * E2 * 2);
    _Float16* ctxWhh16 = (_Float16*)alloc((size_t)2 * G3 * E2 * 2);
    _Float16* modWih16 = (_Float16*)alloc((size_t)2 * G3 * GF * 2);
    _Float16* modWhh16 = (_Float16*)alloc((size_t)2 * G3 * E2 * 2);
    _Float16* p2gWih16 = (_Float16*)alloc((size_t)2 * G3P * E4 * 2);
    _Float16* p2gWhh16 = (_Float16*)alloc((size_t)2 * G3P * E4 * 2);
    _Float16* p1w16    = (_Float16*)alloc((size_t)ANS * 2560 * 2);
    _Float16* p2w16    = (_Float16*)alloc((size_t)ANS * 3072 * 2);
    float*    GI       = (float*)alloc((size_t)B * T * G3P * 4);   // shared gi scratch
    _Float16* hbuf0    = (_Float16*)alloc((size_t)B * E4 * 2);
    _Float16* hbuf1    = (_Float16*)alloc((size_t)B * E4 * 2);
    float*    Cbuf     = (float*)alloc((size_t)B * T * D * 4);
    float*    Qbuf     = (float*)alloc((size_t)B * J * D * 4);
    _Float16* Qpad16   = (_Float16*)alloc((size_t)B * 32 * D * 2);
    _Float16* QT16     = (_Float16*)alloc((size_t)B * D * 32 * 2);
    float*    Cw1      = (float*)alloc((size_t)B * T * 4);
    float*    Qw2      = (float*)alloc((size_t)B * J * 4);
    _Float16* Cw3_16   = (_Float16*)alloc((size_t)B * T * D * 2);
    float*    Sbuf     = (float*)alloc((size_t)B * T * 32 * 4);
    _Float16* S16      = (_Float16*)alloc((size_t)B * T * 32 * 2);
    float*    bwv      = (float*)alloc((size_t)B * T * 4);
    float*    q2cB     = (float*)alloc((size_t)B * D * 4);
    float*    c2qB     = (float*)alloc((size_t)B * T * D * 4);
    float*    Gbuf     = (float*)alloc((size_t)B * T * GF * 4);
    _Float16* G16      = (_Float16*)alloc((size_t)B * T * GF * 2);
    float*    Mbuf     = (float*)alloc((size_t)B * T * D * 4);
    _Float16* M16      = (_Float16*)alloc((size_t)B * T * D * 2);
    float*    M2buf    = (float*)alloc((size_t)B * T * 1024 * 4);
    _Float16* GM16     = (_Float16*)alloc((size_t)B * 2560 * 2);
    _Float16* GM216    = (_Float16*)alloc((size_t)B * 3072 * 2);
    float*    logits   = (float*)alloc((size_t)B * ANS * 4);

    auto ew = [&](long long n) { return dim3((unsigned)((n + 255) / 256)); };
    auto cvt = [&](const float* s, _Float16* d2, long long n) {
        cvt_f32_f16_k<<<ew(n), 256, 0, stream>>>(s, d2, n);
    };
    auto gemm = [&](const _Float16* A, int lda, long long sA,
                    const _Float16* Wt, int ldw, long long sW,
                    const float* bias, float* o, int ldo, long long sO,
                    int M, int N, int K, int batch) {
        dim3 g(M / 16, (N + 127) / 128, batch);
        size_t shbytes = (size_t)(K / 32) * 1280;   // staged A tile (padded rows)
        wmma_gemm_nt<<<g, 256, shbytes, stream>>>(A, lda, sA, Wt, ldw, sW, bias,
                                                  o, ldo, sO, N, K);
    };
    auto scan = [&](const _Float16* Whh16, const float* bhh, int seqLen, int H,
                    float* outAll, int outRow, int dir) {
        (void)hipMemsetAsync(hbuf0, 0, (size_t)B * H * 2, stream);
        _Float16* hc = hbuf0;
        _Float16* hn = hbuf1;
        for (int s = 0; s < seqLen; ++s) {
            int t = dir ? (seqLen - 1 - s) : s;
            gru_step<<<dim3(H / 16), 192, 0, stream>>>(
                hc, hn, Whh16, bhh,
                GI + (long long)t * 3 * H, (long long)seqLen * 3 * H,
                outAll + (long long)t * outRow + (long long)dir * H,
                (long long)seqLen * outRow, H);
            _Float16* tmp = hc; hc = hn; hn = tmp;
        }
    };

    // ---- f16 conversions (weights + inputs) ----
    cvt(embd_ctx, Xc16, (long long)B * T * E2);
    cvt(embd_q,   Xq16, (long long)B * J * E2);
    cvt(ctx_Wih, ctxWih16, 2LL * G3 * E2);
    cvt(ctx_Whh, ctxWhh16, 2LL * G3 * E2);
    cvt(mod_Wih, modWih16, 2LL * G3 * GF);
    cvt(mod_Whh, modWhh16, 2LL * G3 * E2);
    cvt(p2g_Wih, p2gWih16, 2LL * G3P * E4);
    cvt(p2g_Whh, p2gWhh16, 2LL * G3P * E4);
    cvt(p1_w, p1w16, (long long)ANS * 2560);
    cvt(p2_w, p2w16, (long long)ANS * 3072);

    // ---- ctx BiGRU -> Cbuf (B,T,512) ----
    for (int dir = 0; dir < 2; ++dir) {
        gemm(Xc16, E2, 0, ctxWih16 + (long long)dir * G3 * E2, E2, 0,
             ctx_bih + dir * G3, GI, G3, 0, B * T, G3, E2, 1);
        scan(ctxWhh16 + (long long)dir * G3 * E2, ctx_bhh + dir * G3, T, E2, Cbuf, D, dir);
    }
    // ---- query BiGRU -> Qbuf (B,J,512) ----
    for (int dir = 0; dir < 2; ++dir) {
        gemm(Xq16, E2, 0, ctxWih16 + (long long)dir * G3 * E2, E2, 0,
             ctx_bih + dir * G3, GI, G3, 0, B * J, G3, E2, 1);
        scan(ctxWhh16 + (long long)dir * G3 * E2, ctx_bhh + dir * G3, J, E2, Qbuf, D, dir);
    }

    // ---- attention ----
    rowdot_k<<<dim3((B * T + 255) / 256), 256, 0, stream>>>(Cbuf, Wvec,     Cw1, B * T, D);
    rowdot_k<<<dim3((B * J + 255) / 256), 256, 0, stream>>>(Qbuf, Wvec + D, Qw2, B * J, D);
    mulvec_cvt_k<<<ew((long long)B * T * D), 256, 0, stream>>>(Cbuf, Wvec + 2 * D, Cw3_16,
                                                               (long long)B * T * D, D);
    padq_cvt_k<<<ew((long long)B * 32 * D), 256, 0, stream>>>(Qbuf, Qpad16, J, D);
    // S[b,i,j] = (C*w3) @ Q^T  (batched, J padded to 32 with zero rows)
    gemm(Cw3_16, D, (long long)T * D, Qpad16, D, 32LL * D, nullptr,
         Sbuf, 32, (long long)T * 32, T, 32, D, B);
    adds_k<<<ew((long long)B * T * J), 256, 0, stream>>>(Sbuf, Cw1, Qw2, T, J);
    cvt(Sbuf, S16, (long long)B * T * 32);
    transq_k<<<ew((long long)B * D * 32), 256, 0, stream>>>(Qpad16, QT16, D);
    // c2q = S @ Q  (padded columns hit zero Q rows)
    gemm(S16, 32, (long long)T * 32, QT16, 32, (long long)D * 32, nullptr,
         c2qB, D, (long long)T * D, T, D, 32, B);
    attn_b_softmax_k<<<dim3(B), 256, 0, stream>>>(Sbuf, bwv, T, J);
    q2c_k<<<dim3(B), D, 0, stream>>>(bwv, Cbuf, q2cB, T, D);
    build_g_k<<<ew((long long)B * T * GF), 256, 0, stream>>>(Cbuf, c2qB, q2cB, Gbuf, G16, T, D);

    // ---- mod BiGRU: G -> Mbuf (B,T,512) ----
    for (int dir = 0; dir < 2; ++dir) {
        gemm(G16, GF, 0, modWih16 + (long long)dir * G3 * GF, GF, 0,
             mod_bih + dir * G3, GI, G3, 0, B * T, G3, GF, 1);
        scan(modWhh16 + (long long)dir * G3 * E2, mod_bhh + dir * G3, T, E2, Mbuf, D, dir);
    }
    // ---- head 1 ----
    gm_sum_k<<<ew((long long)B * 2560), 256, 0, stream>>>(Gbuf, Mbuf, GM16, T, GF, D);
    gemm(GM16, 2560, 0, p1w16, 2560, 0, p1_b, logits, ANS, 0, B, ANS, 2560, 1);
    softmax_rows_k<<<dim3(B), 256, 0, stream>>>(logits, out, ANS);

    // ---- p2g BiGRU: M -> M2buf (B,T,1024) ----
    cvt(Mbuf, M16, (long long)B * T * D);
    for (int dir = 0; dir < 2; ++dir) {
        gemm(M16, D, 0, p2gWih16 + (long long)dir * G3P * E4, D, 0,
             p2g_bih + dir * G3P, GI, G3P, 0, B * T, G3P, D, 1);
        scan(p2gWhh16 + (long long)dir * G3P * E4, p2g_bhh + dir * G3P, T, E4, M2buf, 1024, dir);
    }
    // ---- head 2 ----
    gm_sum_k<<<ew((long long)B * 3072), 256, 0, stream>>>(Gbuf, M2buf, GM216, T, GF, 1024);
    gemm(GM216, 3072, 0, p2w16, 3072, 0, p2_b, logits, ANS, 0, B, ANS, 3072, 1);
    softmax_rows_k<<<dim3(B), 256, 0, stream>>>(logits, out + B * ANS, ANS);
}
